// GNN_LSTM_8160437863163
// MI455X (gfx1250) — compile-verified, matrix operand order checked by reading, not run on hardware
//
#include <hip/hip_runtime.h>
#include <hip/hip_bf16.h>
#include <math.h>

// Problem constants (match reference)
#define TT   4
#define NN   20000
#define EE   320000
#define FF   64
#define HH   64
#define TSN  50
#define KSEL 10000          // max(1, int(N*0.5))
#define NH   (NN * HH)      // 1,280,000

typedef __attribute__((ext_vector_type(2)))  float    v2f;
typedef __attribute__((ext_vector_type(8)))  float    v8f;
typedef __attribute__((ext_vector_type(16))) _Float16 v16h;

#ifdef __has_builtin
#if __has_builtin(__builtin_amdgcn_wmma_f32_16x16x4_f32)
#define HAVE_WMMA_F32X4 1
#endif
#endif

// ---------------------------------------------------------------- utilities
__global__ void gl_zero(float* p, int n) {
    int i = blockIdx.x * blockDim.x + threadIdx.x;
    if (i < n) p[i] = 0.0f;
}
__global__ void gl_copy(float* dst, const float* src, int n) {
    int i = blockIdx.x * blockDim.x + threadIdx.x;
    if (i < n) dst[i] = src[i];
}

// ---------------------------------------------------------------- GCN norm precompute
__global__ void gl_deg_scatter(const long long* __restrict__ ei, float* __restrict__ deg) {
    int e = blockIdx.x * blockDim.x + threadIdx.x;
    if (e < EE) {
        int dst = (int)ei[EE + e];
        atomicAdd(&deg[dst], 1.0f);
    }
}
__global__ void gl_deg_final(float* __restrict__ deg, float* __restrict__ dinv,
                             float* __restrict__ selfn) {
    int i = blockIdx.x * blockDim.x + threadIdx.x;
    if (i < NN) {
        float d = deg[i] + 1.0f;
        deg[i]   = d;
        dinv[i]  = rsqrtf(d);
        selfn[i] = 1.0f / d;
    }
}
__global__ void gl_gcn_norm(const long long* __restrict__ ei, const float* __restrict__ dinv,
                            float* __restrict__ gcn) {
    int e = blockIdx.x * blockDim.x + threadIdx.x;
    if (e < EE) {
        int src = (int)ei[e];
        int dst = (int)ei[EE + e];
        gcn[e] = dinv[src] * dinv[dst];
    }
}

// ---------------------------------------------------------------- per-step standardize x
__global__ void gl_colstats(const float* __restrict__ x, float* __restrict__ mean,
                            float* __restrict__ rstd) {
    __shared__ float s1[256], s2[256];
    int f = blockIdx.x;           // feature 0..63
    float a = 0.f, b = 0.f;
    for (int i = threadIdx.x; i < NN; i += 256) {
        float v = x[(size_t)i * FF + f];
        a += v; b += v * v;
    }
    s1[threadIdx.x] = a; s2[threadIdx.x] = b;
    __syncthreads();
    for (int s = 128; s > 0; s >>= 1) {
        if (threadIdx.x < s) { s1[threadIdx.x] += s1[threadIdx.x + s];
                               s2[threadIdx.x] += s2[threadIdx.x + s]; }
        __syncthreads();
    }
    if (threadIdx.x == 0) {
        float mu  = s1[0] / (float)NN;
        float var = (s2[0] - (float)NN * mu * mu) / (float)(NN - 1);   // unbiased (torch std)
        var = var > 0.f ? var : 0.f;
        mean[f] = mu;
        rstd[f] = 1.0f / (sqrtf(var) + 1e-6f);
    }
}
__global__ void gl_xnorm(const float* __restrict__ x, const float* __restrict__ mean,
                         const float* __restrict__ rstd, float* __restrict__ xn) {
    int idx = blockIdx.x * blockDim.x + threadIdx.x;
    if (idx < NH) {
        int f = idx & 63;
        xn[idx] = (x[idx] - mean[f]) * rstd[f];
    }
}

// ---------------------------------------------------------------- sparse aggregations
__global__ void gl_aggh_init(const float* __restrict__ h, const float* __restrict__ selfn,
                             float* __restrict__ aggh) {
    int idx = blockIdx.x * blockDim.x + threadIdx.x;
    if (idx < NH) aggh[idx] = selfn[idx >> 6] * h[idx];
}
__global__ void gl_scatter_x(const long long* __restrict__ ei, const float* __restrict__ w,
                             const float* __restrict__ xn, float* __restrict__ aggx) {
    long long idx = (long long)blockIdx.x * blockDim.x + threadIdx.x;
    if (idx < (long long)EE * FF) {
        int e = (int)(idx >> 6), f = (int)(idx & 63);
        int src = (int)ei[e], dst = (int)ei[EE + e];
        atomicAdd(&aggx[(size_t)dst * FF + f], w[e] * xn[(size_t)src * FF + f]);
    }
}
__global__ void gl_scatter_h(const long long* __restrict__ ei, const float* __restrict__ gcn,
                             const float* __restrict__ h, float* __restrict__ aggh) {
    long long idx = (long long)blockIdx.x * blockDim.x + threadIdx.x;
    if (idx < (long long)EE * HH) {
        int e = (int)(idx >> 6), f = (int)(idx & 63);
        int src = (int)ei[e], dst = (int)ei[EE + e];
        atomicAdd(&aggh[(size_t)dst * HH + f], gcn[e] * h[(size_t)src * HH + f]);
    }
}

// ---------------------------------------------------------------- fused gate GEMM (WMMA)
// gates[g][n][h] = sum_k A[n,k]*B_g[k,h],  A = [agg_x | x_norm | agg_h]  (K = 192)
// One wave per (16-row tile, gate); 4 col-tiles of 16 -> full H=64.
__global__ __launch_bounds__(128)
void gl_gates_wmma(const float* __restrict__ aggx, const float* __restrict__ xn,
                   const float* __restrict__ aggh,
                   const float* __restrict__ Wrel, const float* __restrict__ Wroot,
                   const float* __restrict__ Wg, float* __restrict__ gates) {
    const int lane = threadIdx.x & 31;
    const int g    = threadIdx.x >> 5;     // gate 0..3 (one wave each)
    const int rt   = blockIdx.x;           // row tile 0..1249 (20000/16 exact)
    const int m    = lane & 15;
    const int half = lane >> 4;            // 0 or 1
    const int row  = rt * 16 + m;
    const int n0   = lane & 15;

    v8f acc0 = {0,0,0,0,0,0,0,0}, acc1 = {0,0,0,0,0,0,0,0};
    v8f acc2 = {0,0,0,0,0,0,0,0}, acc3 = {0,0,0,0,0,0,0,0};

#if HAVE_WMMA_F32X4
    const int kh = half * 2;               // A: lanes 0-15 hold K=0,1; lanes 16-31 hold K=2,3
    for (int kb = 0; kb < 192; kb += 4) {
        const int reg = kb >> 6;
        const float* A = (reg == 0) ? aggx : (reg == 1) ? xn : aggh;
        const float* B = (reg == 0) ? Wrel : (reg == 1) ? Wroot : Wg;
        const int ks = kb & 63;
        v2f a;
        a.x = A[(size_t)row * 64 + ks + kh];
        a.y = A[(size_t)row * 64 + ks + kh + 1];
        const float* Bg = B + (size_t)g * 4096 + (size_t)(ks + kh) * 64;
        v2f b0, b1, b2, b3;
        b0.x = Bg[n0];        b0.y = Bg[64 + n0];
        b1.x = Bg[16 + n0];   b1.y = Bg[80 + n0];
        b2.x = Bg[32 + n0];   b2.y = Bg[96 + n0];
        b3.x = Bg[48 + n0];   b3.y = Bg[112 + n0];
        acc0 = __builtin_amdgcn_wmma_f32_16x16x4_f32(false, a, false, b0, (short)0, acc0, false, false);
        acc1 = __builtin_amdgcn_wmma_f32_16x16x4_f32(false, a, false, b1, (short)0, acc1, false, false);
        acc2 = __builtin_amdgcn_wmma_f32_16x16x4_f32(false, a, false, b2, (short)0, acc2, false, false);
        acc3 = __builtin_amdgcn_wmma_f32_16x16x4_f32(false, a, false, b3, (short)0, acc3, false, false);
    }
#else
    // Fallback: f16 inputs, f32 accumulate (codegen-confirmed builtin), K step 32
    for (int kb = 0; kb < 192; kb += 32) {
        const int reg = kb >> 6;
        const float* A = (reg == 0) ? aggx : (reg == 1) ? xn : aggh;
        const float* B = (reg == 0) ? Wrel : (reg == 1) ? Wroot : Wg;
        const int ks = kb & 63;
        v16h a, b0, b1, b2, b3;
        #pragma unroll
        for (int p = 0; p < 8; ++p) {
            int kl = ((p < 4) ? (p * 2) : (16 + (p - 4) * 2)) + half * 8;
            a[2*p]   = (_Float16)A[(size_t)row * 64 + ks + kl];
            a[2*p+1] = (_Float16)A[(size_t)row * 64 + ks + kl + 1];
            const float* Br0 = B + (size_t)g * 4096 + (size_t)(ks + kl) * 64;
            const float* Br1 = Br0 + 64;
            b0[2*p] = (_Float16)Br0[n0];      b0[2*p+1] = (_Float16)Br1[n0];
            b1[2*p] = (_Float16)Br0[16+n0];   b1[2*p+1] = (_Float16)Br1[16+n0];
            b2[2*p] = (_Float16)Br0[32+n0];   b2[2*p+1] = (_Float16)Br1[32+n0];
            b3[2*p] = (_Float16)Br0[48+n0];   b3[2*p+1] = (_Float16)Br1[48+n0];
        }
        acc0 = __builtin_amdgcn_wmma_f32_16x16x32_f16(false, a, false, b0, (short)0, acc0, false, false);
        acc1 = __builtin_amdgcn_wmma_f32_16x16x32_f16(false, a, false, b1, (short)0, acc1, false, false);
        acc2 = __builtin_amdgcn_wmma_f32_16x16x32_f16(false, a, false, b2, (short)0, acc2, false, false);
        acc3 = __builtin_amdgcn_wmma_f32_16x16x32_f16(false, a, false, b3, (short)0, acc3, false, false);
    }
#endif
    // C/D layout: VGPR v -> row (v + 8*half), col = lane&15 (+16*coltile)
    float* Gp = gates + (size_t)g * NH;
    const int rb = half * 8;
    #pragma unroll
    for (int v = 0; v < 8; ++v) {
        const int rg = rt * 16 + rb + v;
        float* p = Gp + (size_t)rg * 64 + n0;
        p[0]  = acc0[v];
        p[16] = acc1[v];
        p[32] = acc2[v];
        p[48] = acc3[v];
    }
}

// ---------------------------------------------------------------- cell/hidden update
__global__ void gl_cell_update(const float* __restrict__ gates, const float* __restrict__ brel,
                               const float* __restrict__ bg, float* __restrict__ h,
                               float* __restrict__ c) {
    int idx = blockIdx.x * blockDim.x + threadIdx.x;
    if (idx >= NH) return;
    int j = idx & 63;
    float g0 = gates[idx]              + brel[j]       + bg[j];
    float g1 = gates[(size_t)NH + idx] + brel[64 + j]  + bg[64 + j];
    float g2 = gates[2*(size_t)NH+idx] + brel[128 + j] + bg[128 + j];
    float g3 = gates[3*(size_t)NH+idx] + brel[192 + j] + bg[192 + j];
    float ig = 1.0f / (1.0f + expf(-g0));
    float fg = 1.0f / (1.0f + expf(-g1));
    float og = 1.0f / (1.0f + expf(-g2));
    float mg = g3 > 0.f ? g3 : 0.f;
    float cn = tanhf(ig * mg + fg * c[idx]);
    c[idx] = cn;
    h[idx] = og * tanhf(cn);
}

// ---------------------------------------------------------------- DGPool
__global__ void gl_vnorm(const float* __restrict__ pv, float* __restrict__ v) {
    __shared__ float s[64];
    int j = threadIdx.x;
    float x = pv[j];
    s[j] = x * x;
    __syncthreads();
    for (int k = 32; k > 0; k >>= 1) { if (j < k) s[j] += s[j + k]; __syncthreads(); }
    v[j] = x / (sqrtf(s[0]) + 1e-8f);
}
__global__ void gl_scores(const float* __restrict__ h, const float* __restrict__ v,
                          float* __restrict__ scores, float* __restrict__ red) {
    __shared__ float s1[256], s2[256];
    int i = blockIdx.x * blockDim.x + threadIdx.x;
    float sc = 0.f;
    if (i < NN) {
        const float* hr = h + (size_t)i * HH;
        #pragma unroll
        for (int j = 0; j < HH; ++j) sc += hr[j] * v[j];
        scores[i] = sc;
    }
    s1[threadIdx.x] = (i < NN) ? sc : 0.f;
    s2[threadIdx.x] = (i < NN) ? sc * sc : 0.f;
    __syncthreads();
    for (int k = 128; k > 0; k >>= 1) {
        if (threadIdx.x < k) { s1[threadIdx.x] += s1[threadIdx.x + k];
                               s2[threadIdx.x] += s2[threadIdx.x + k]; }
        __syncthreads();
    }
    if (threadIdx.x == 0) { atomicAdd(&red[0], s1[0]); atomicAdd(&red[1], s2[0]); }
}
__global__ void gl_score_stats(const float* __restrict__ red, float* __restrict__ muinv) {
    float mu  = red[0] / (float)NN;
    float var = red[1] / (float)NN - mu * mu;       // biased (jnp .std default)
    var = var > 0.f ? var : 0.f;
    muinv[0] = mu;
    muinv[1] = 1.0f / (sqrtf(var) + 1e-8f);
}
__global__ void gl_score_std(const float* __restrict__ scores, const float* __restrict__ muinv,
                             float* __restrict__ out_scores, float* __restrict__ sig,
                             unsigned* __restrict__ keys, float* __restrict__ plred) {
    __shared__ float s[256];
    int i = blockIdx.x * blockDim.x + threadIdx.x;
    float pl = 0.f;
    if (i < NN) {
        float ss = (scores[i] - muinv[0]) * muinv[1];
        out_scores[i] = ss;
        float sg = 1.0f / (1.0f + expf(-ss));
        sig[i] = sg;
        unsigned u = __float_as_uint(ss);
        keys[i] = (u & 0x80000000u) ? ~u : (u | 0x80000000u);  // monotonic key
        pl = sg * (1.0f - sg);
    }
    s[threadIdx.x] = pl;
    __syncthreads();
    for (int k = 128; k > 0; k >>= 1) {
        if (threadIdx.x < k) s[threadIdx.x] += s[threadIdx.x + k];
        __syncthreads();
    }
    if (threadIdx.x == 0) atomicAdd(plred, s[0]);
}
// st: [0]=lo [1]=hi [2]=mid [3]=counter
__global__ void gl_bisect_init(unsigned* st) { st[0] = 0u; st[1] = 0xFFFFFFFFu; st[2] = 0u; st[3] = 0u; }
__global__ void gl_bisect_pre(unsigned* st) {
    unsigned long long lo = st[0], hi = st[1];
    st[2] = (unsigned)(lo + ((hi - lo + 1ull) >> 1));
    st[3] = 0u;
}
__global__ void gl_count_ge(const unsigned* __restrict__ keys, unsigned* __restrict__ st) {
    __shared__ unsigned s[256];
    int i = blockIdx.x * blockDim.x + threadIdx.x;
    unsigned mid = st[2];
    s[threadIdx.x] = (i < NN && keys[i] >= mid) ? 1u : 0u;
    __syncthreads();
    for (int k = 128; k > 0; k >>= 1) {
        if (threadIdx.x < k) s[threadIdx.x] += s[threadIdx.x + k];
        __syncthreads();
    }
    if (threadIdx.x == 0) atomicAdd(&st[3], s[0]);
}
__global__ void gl_bisect_post(unsigned* st) {
    if (st[3] >= (unsigned)KSEL) st[0] = st[2];
    else st[1] = st[2] - 1u;
}
__global__ void gl_select(const unsigned* __restrict__ keys, const unsigned* __restrict__ st,
                          const float* __restrict__ h, const float* __restrict__ sig,
                          float* __restrict__ high, unsigned* __restrict__ selcnt) {
    int idx = blockIdx.x * blockDim.x + threadIdx.x;
    if (idx >= NH) return;
    int i = idx >> 6, j = idx & 63;
    if (keys[i] >= st[0]) {
        atomicAdd(&high[j], h[(size_t)i * HH + j] * sig[i]);
        if (j == 0) atomicAdd(selcnt, 1u);
    }
}
__global__ void gl_high_fin(float* __restrict__ high, const unsigned* __restrict__ selcnt) {
    int j = threadIdx.x;
    unsigned c = *selcnt; if (c == 0u) c = 1u;
    high[j] /= (float)c;
}

// ---------------------------------------------------------------- raw-fMRI LSTM
__global__ void gl_gih(const float* __restrict__ W_ih, const float* __restrict__ xt,
                       float* __restrict__ gih) {
    __shared__ float s[256];
    int r = blockIdx.x;                       // output row 0..255
    const float* Wr = W_ih + (size_t)r * NN;
    float a = 0.f;
    for (int i = threadIdx.x; i < NN; i += 256) a += Wr[i] * xt[i];
    s[threadIdx.x] = a;
    __syncthreads();
    for (int k = 128; k > 0; k >>= 1) {
        if (threadIdx.x < k) s[threadIdx.x] += s[threadIdx.x + k];
        __syncthreads();
    }
    if (threadIdx.x == 0) gih[r] = s[0];
}
__global__ void gl_lstm_combine(const float* __restrict__ gih, const float* __restrict__ b_ih,
                                const float* __restrict__ b_hh, const float* __restrict__ W_hh,
                                float* __restrict__ hl, float* __restrict__ cl) {
    __shared__ float hs[64], gs[256];
    int r = threadIdx.x;
    if (r < 64) hs[r] = hl[r];
    __syncthreads();
    float g = gih[r] + b_ih[r] + b_hh[r];
    const float* Wr = W_hh + (size_t)r * HH;
    #pragma unroll
    for (int j = 0; j < HH; ++j) g += Wr[j] * hs[j];
    gs[r] = g;
    __syncthreads();
    if (r < 64) {
        float gi = 1.0f / (1.0f + expf(-gs[r]));
        float gf = 1.0f / (1.0f + expf(-gs[64 + r]));
        float gg = tanhf(gs[128 + r]);
        float go = 1.0f / (1.0f + expf(-gs[192 + r]));
        float c  = gf * cl[r] + gi * gg;
        cl[r] = c;
        hl[r] = go * tanhf(c);
    }
}

// ---------------------------------------------------------------- head: LN + MLP
__global__ void gl_head(const float* __restrict__ high, const float* __restrict__ hl,
                        const float* __restrict__ ln_g, const float* __restrict__ ln_b,
                        const float* __restrict__ W1, const float* __restrict__ b1,
                        const float* __restrict__ W2, const float* __restrict__ b2,
                        const float* __restrict__ plred, float* __restrict__ d_out) {
    __shared__ float fus[128], red[128], z[64];
    int t = threadIdx.x;                     // 0..127
    float x = (t < 64) ? high[t] : hl[t - 64];
    fus[t] = x;
    red[t] = x;
    __syncthreads();
    for (int k = 64; k > 0; k >>= 1) { if (t < k) red[t] += red[t + k]; __syncthreads(); }
    float mu = red[0] / 128.0f;
    __syncthreads();
    float d = fus[t] - mu;
    red[t] = d * d;
    __syncthreads();
    for (int k = 64; k > 0; k >>= 1) { if (t < k) red[t] += red[t + k]; __syncthreads(); }
    float var = red[0] / 128.0f;
    float rs = rsqrtf(var + 1e-5f);
    __syncthreads();
    fus[t] = (fus[t] - mu) * rs * ln_g[t] + ln_b[t];
    __syncthreads();
    if (t < 64) {
        float a = b1[t];
        #pragma unroll
        for (int i = 0; i < 128; ++i) a += fus[i] * W1[(size_t)i * 64 + t];
        z[t] = a > 0.f ? a : 0.f;
    }
    __syncthreads();
    if (t == 0) {
        float p = b2[0];
        for (int j = 0; j < 64; ++j) p += z[j] * W2[j];
        d_out[0] = p;                         // pred
        d_out[1 + NN] = plred[0] / (float)NN; // pool_loss
    }
}

// ================================================================ launch
extern "C" void kernel_launch(void* const* d_in, const int* in_sizes, int n_in,
                              void* d_out, int out_size, void* d_ws, size_t ws_size,
                              hipStream_t stream) {
    const float*     xs    = (const float*)d_in[0];
    const long long* ei    = (const long long*)d_in[1];
    const float*     ea    = (const float*)d_in[2];
    const float*     h0    = (const float*)d_in[3];
    const float*     c0    = (const float*)d_in[4];
    const float*     ts    = (const float*)d_in[5];
    const float*     Wrel  = (const float*)d_in[6];
    const float*     brel  = (const float*)d_in[7];
    const float*     Wroot = (const float*)d_in[8];
    const float*     Wg    = (const float*)d_in[9];
    const float*     bg    = (const float*)d_in[10];
    const float*     pv    = (const float*)d_in[11];
    const float*     ln_g  = (const float*)d_in[12];
    const float*     ln_b  = (const float*)d_in[13];
    const float*     W_ih  = (const float*)d_in[14];
    const float*     W_hh  = (const float*)d_in[15];
    const float*     b_ih  = (const float*)d_in[16];
    const float*     b_hh  = (const float*)d_in[17];
    const float*     W1    = (const float*)d_in[18];
    const float*     b1    = (const float*)d_in[19];
    const float*     W2    = (const float*)d_in[20];
    const float*     b2    = (const float*)d_in[21];
    float* out = (float*)d_out;

    // workspace carve-out
    char* base = (char*)d_ws;
    auto carve = [&](size_t bytes) { char* p = base; base += (bytes + 255) & ~(size_t)255; return p; };
    float* xnorm  = (float*)carve(sizeof(float) * NH);
    float* aggx   = (float*)carve(sizeof(float) * NH);
    float* aggh   = (float*)carve(sizeof(float) * NH);
    float* gates  = (float*)carve(sizeof(float) * 4 * (size_t)NH);
    float* hbuf   = (float*)carve(sizeof(float) * NH);
    float* cbuf   = (float*)carve(sizeof(float) * NH);
    float* deg    = (float*)carve(sizeof(float) * NN);
    float* dinv   = (float*)carve(sizeof(float) * NN);
    float* selfn  = (float*)carve(sizeof(float) * NN);
    float* gcn    = (float*)carve(sizeof(float) * EE);
    float* cmean  = (float*)carve(sizeof(float) * FF);
    float* crstd  = (float*)carve(sizeof(float) * FF);
    float* vvec   = (float*)carve(sizeof(float) * HH);
    float* scores = (float*)carve(sizeof(float) * NN);
    float* sig    = (float*)carve(sizeof(float) * NN);
    unsigned* keys   = (unsigned*)carve(sizeof(unsigned) * NN);
    float* red    = (float*)carve(sizeof(float) * 4);     // [sum, sumsq, plAcc, spare]
    float* muinv  = (float*)carve(sizeof(float) * 2);
    unsigned* bst    = (unsigned*)carve(sizeof(unsigned) * 4);
    float* high   = (float*)carve(sizeof(float) * HH);
    unsigned* selcnt = (unsigned*)carve(sizeof(unsigned) * 1);
    float* gih    = (float*)carve(sizeof(float) * 256);
    float* hl     = (float*)carve(sizeof(float) * HH);
    float* cl     = (float*)carve(sizeof(float) * HH);
    (void)ws_size; (void)n_in; (void)in_sizes; (void)out_size;

    const int TPB = 256;
    const int gN   = (NN + TPB - 1) / TPB;
    const int gNH  = (NH + TPB - 1) / TPB;
    const int gE   = (EE + TPB - 1) / TPB;
    const int gEF  = (int)(((long long)EE * FF + TPB - 1) / TPB);

    // ---- precompute GCN normalization
    gl_zero<<<gN, TPB, 0, stream>>>(deg, NN);
    gl_deg_scatter<<<gE, TPB, 0, stream>>>(ei, deg);
    gl_deg_final<<<gN, TPB, 0, stream>>>(deg, dinv, selfn);
    gl_gcn_norm<<<gE, TPB, 0, stream>>>(ei, dinv, gcn);

    // ---- init recurrent state
    gl_copy<<<gNH, TPB, 0, stream>>>(hbuf, h0, NH);
    gl_copy<<<gNH, TPB, 0, stream>>>(cbuf, c0, NH);

    // ---- GNN recurrence over T timesteps
    for (int t = 0; t < TT; ++t) {
        const float* xt = xs + (size_t)t * NH;
        const float* wt = ea + (size_t)t * EE;
        gl_colstats<<<FF, TPB, 0, stream>>>(xt, cmean, crstd);
        gl_xnorm<<<gNH, TPB, 0, stream>>>(xt, cmean, crstd, xnorm);
        gl_zero<<<gNH, TPB, 0, stream>>>(aggx, NH);
        gl_scatter_x<<<gEF, TPB, 0, stream>>>(ei, wt, xnorm, aggx);
        gl_aggh_init<<<gNH, TPB, 0, stream>>>(hbuf, selfn, aggh);
        gl_scatter_h<<<gEF, TPB, 0, stream>>>(ei, gcn, hbuf, aggh);
        gl_gates_wmma<<<NN / 16, 128, 0, stream>>>(aggx, xnorm, aggh, Wrel, Wroot, Wg, gates);
        gl_cell_update<<<gNH, TPB, 0, stream>>>(gates, brel, bg, hbuf, cbuf);
    }

    // ---- DGPool
    gl_vnorm<<<1, 64, 0, stream>>>(pv, vvec);
    gl_zero<<<1, 4, 0, stream>>>(red, 4);
    gl_scores<<<gN, TPB, 0, stream>>>(hbuf, vvec, scores, red);
    gl_score_stats<<<1, 1, 0, stream>>>(red, muinv);
    gl_score_std<<<gN, TPB, 0, stream>>>(scores, muinv, out + 1, sig, keys, &red[2]);

    gl_bisect_init<<<1, 1, 0, stream>>>(bst);
    for (int it = 0; it < 32; ++it) {
        gl_bisect_pre<<<1, 1, 0, stream>>>(bst);
        gl_count_ge<<<gN, TPB, 0, stream>>>(keys, bst);
        gl_bisect_post<<<1, 1, 0, stream>>>(bst);
    }
    gl_zero<<<1, 64, 0, stream>>>(high, HH);
    gl_zero<<<1, 1, 0, stream>>>((float*)selcnt, 1);
    gl_select<<<gNH, TPB, 0, stream>>>(keys, bst, hbuf, sig, high, selcnt);
    gl_high_fin<<<1, 64, 0, stream>>>(high, selcnt);

    // ---- raw-fMRI LSTM over TS steps
    gl_zero<<<1, 64, 0, stream>>>(hl, HH);
    gl_zero<<<1, 64, 0, stream>>>(cl, HH);
    for (int t = 0; t < TSN; ++t) {
        gl_gih<<<256, TPB, 0, stream>>>(W_ih, ts + (size_t)t * NN, gih);
        gl_lstm_combine<<<1, 256, 0, stream>>>(gih, b_ih, b_hh, W_hh, hl, cl);
    }

    // ---- fusion LayerNorm + MLP head
    gl_head<<<1, 128, 0, stream>>>(high, hl, ln_g, ln_b, W1, b1, W2, b2, &red[2], out);
}
